// DepthSegmNetAttention06_25752623907060
// MI455X (gfx1250) — matrix operand order repaired and checked by reading, not compiled, not run on hardware
//
#include <hip/hip_runtime.h>
#include <hip/hip_bf16.h>

typedef __attribute__((ext_vector_type(16))) _Float16 v16h;
typedef __attribute__((ext_vector_type(8)))  _Float16 v8h;
typedef __attribute__((ext_vector_type(8)))  float    v8f;

#define HSZ 96
#define NHEAD 3
#define HDIM 32
#define SEQ 1024
#define NBATCH 16
#define BSROWS (NBATCH * SEQ)   /* 16384 rows in the flattened activation */
#define MLPD 1024
#define NLAYER 3

// ---------------------------------------------------------------------------
// wave32 reductions within 16-lane half-groups (rows of a C fragment)
// ---------------------------------------------------------------------------
__device__ __forceinline__ float hred_max16(float t) {
  t = fmaxf(t, __shfl_xor(t, 1, 32));
  t = fmaxf(t, __shfl_xor(t, 2, 32));
  t = fmaxf(t, __shfl_xor(t, 4, 32));
  t = fmaxf(t, __shfl_xor(t, 8, 32));
  return t;
}
__device__ __forceinline__ float hred_sum16(float t) {
  t += __shfl_xor(t, 1, 32);
  t += __shfl_xor(t, 2, 32);
  t += __shfl_xor(t, 4, 32);
  t += __shfl_xor(t, 8, 32);
  return t;
}

// ---------------------------------------------------------------------------
// LayerNorm: one wave per row of width 96. OUT32: write f32, else f16.
// ---------------------------------------------------------------------------
template <bool OUT32>
__global__ void ln_kernel(const float* __restrict__ x,
                          const float* __restrict__ g,
                          const float* __restrict__ b,
                          void* __restrict__ out, int rows) {
  int wid  = (blockIdx.x * blockDim.x + threadIdx.x) >> 5;
  int lane = threadIdx.x & 31;
  if (wid >= rows) return;
  const float* row = x + (size_t)wid * HSZ;
  float v0 = row[lane], v1 = row[lane + 32], v2 = row[lane + 64];
  float s = v0 + v1 + v2;
  s += __shfl_xor(s, 1, 32);  s += __shfl_xor(s, 2, 32);
  s += __shfl_xor(s, 4, 32);  s += __shfl_xor(s, 8, 32);
  s += __shfl_xor(s, 16, 32);
  float mu = s * (1.0f / 96.0f);
  float d0 = v0 - mu, d1 = v1 - mu, d2 = v2 - mu;
  float vs = d0 * d0 + d1 * d1 + d2 * d2;
  vs += __shfl_xor(vs, 1, 32);  vs += __shfl_xor(vs, 2, 32);
  vs += __shfl_xor(vs, 4, 32);  vs += __shfl_xor(vs, 8, 32);
  vs += __shfl_xor(vs, 16, 32);
  float inv = rsqrtf(vs * (1.0f / 96.0f) + 1e-6f);
  float o0 = d0 * inv * g[lane]      + b[lane];
  float o1 = d1 * inv * g[lane + 32] + b[lane + 32];
  float o2 = d2 * inv * g[lane + 64] + b[lane + 64];
  if constexpr (OUT32) {
    float* op = (float*)out + (size_t)wid * HSZ;
    op[lane] = o0; op[lane + 32] = o1; op[lane + 64] = o2;
  } else {
    _Float16* op = (_Float16*)out + (size_t)wid * HSZ;
    op[lane] = (_Float16)o0; op[lane + 32] = (_Float16)o1; op[lane + 64] = (_Float16)o2;
  }
}

// ---------------------------------------------------------------------------
// Pack f32 weights [K,N] row-major into WMMA B-fragment order (f16):
// block (kt,nt) of 512 halves; index lane*16+h  ->  W[kt*32 + (lane&16?16+h:h)]
//                                                   [nt*16 + (lane&15)]
// blockIdx.y = layer (weight tensors are [L,K,N]).
// ---------------------------------------------------------------------------
__global__ void pack_w_kernel(const float* __restrict__ W,
                              _Float16* __restrict__ out, int K, int N) {
  int layer  = blockIdx.y;
  int tid    = blockIdx.x * 256 + threadIdx.x;
  int total  = K * N;
  if (tid >= total) return;
  int ntiles = N >> 4;
  int tile = tid >> 9, e = tid & 511;
  int lane = e >> 4,  h = e & 15;
  int kt = tile / ntiles, nt = tile % ntiles;
  int k  = kt * 32 + ((lane & 16) ? 16 + h : h);
  int n  = nt * 16 + (lane & 15);
  out[(size_t)layer * total + tid] =
      (_Float16)W[(size_t)layer * total + (size_t)k * N + n];
}

// ---------------------------------------------------------------------------
// Transpose projected V [BS,HS] (f16, head-interleaved) -> VT [(b,h,d)][s]
// so attention B-fragments for V are contiguous 32B loads.
// ---------------------------------------------------------------------------
__global__ void vtrans_kernel(const _Float16* __restrict__ Vp,
                              _Float16* __restrict__ VT) {
  size_t idx = (size_t)blockIdx.x * 256 + threadIdx.x;
  if (idx >= (size_t)NBATCH * NHEAD * HDIM * SEQ) return;
  int s = (int)(idx & (SEQ - 1));
  size_t t = idx >> 10;
  int d = (int)(t & (HDIM - 1));  t >>= 5;
  int h = (int)(t % NHEAD);
  int b = (int)(t / NHEAD);
  VT[idx] = Vp[((size_t)b * SEQ + s) * HSZ + h * HDIM + d];
}

// ---------------------------------------------------------------------------
// GEMM: out[M,N] = act[M,K](f16) x Wpacked[K,N](f16 B-frag order) + bias
// One wave computes a (16*MT) x (16*NT) tile: MT A-fragments reuse each of
// the NT B-fragments. MT*NT is kept small enough (<=8 accumulators) that the
// hoisted load clauses never spill accumulators to scratch.
// ---------------------------------------------------------------------------
template <int MT, int NT, int NDIM, int KDIM, bool RELU, bool RES, bool OUT32>
__global__ void gemm_kernel(const _Float16* __restrict__ A,
                            const _Float16* __restrict__ Bw,
                            const float* __restrict__ bias,
                            const float* __restrict__ res,
                            void* __restrict__ out, int M) {
  constexpr int NTILES  = NDIM / 16;
  constexpr int NGROUPS = NDIM / (16 * NT);
  constexpr int KITER   = KDIM / 32;
  int lane = threadIdx.x & 31;
  int gid  = (blockIdx.x * blockDim.x + threadIdx.x) >> 5;
  int mt = gid / NGROUPS, ng = gid % NGROUPS;
  if (mt >= M / (16 * MT)) return;     // wave-uniform
  int row0 = mt * (16 * MT), col0 = ng * (16 * NT);

  v8f zero = {};
  v8f acc[MT * NT];
#pragma unroll
  for (int i = 0; i < MT * NT; ++i) acc[i] = zero;

  const _Float16* ap =
      A + (size_t)(row0 + (lane & 15)) * KDIM + ((lane & 16) ? 16 : 0);
  const _Float16* bp =
      Bw + (size_t)(col0 >> 4) * 512 + lane * 16;

#pragma unroll 4
  for (int kt = 0; kt < KITER; ++kt) {
    v16h af[MT];
#pragma unroll
    for (int mi = 0; mi < MT; ++mi) {
      const _Float16* api = ap + (size_t)mi * 16 * KDIM;
      v8h alo = *(const v8h*)(api);
      v8h ahi = *(const v8h*)(api + 8);
#pragma unroll
      for (int i = 0; i < 8; ++i) { af[mi][i] = alo[i]; af[mi][i + 8] = ahi[i]; }
    }
    v16h bfr[NT];
#pragma unroll
    for (int nt = 0; nt < NT; ++nt)
      bfr[nt] = *(const v16h*)(bp + nt * 512);
#pragma unroll
    for (int nt = 0; nt < NT; ++nt)
#pragma unroll
      for (int mi = 0; mi < MT; ++mi)
        acc[mi * NT + nt] = __builtin_amdgcn_wmma_f32_16x16x32_f16(
            false, af[mi], false, bfr[nt], (short)0, acc[mi * NT + nt],
            false, false);
    ap += 32;
    bp += (size_t)NTILES * 512;
  }

  int half = lane >> 4;
#pragma unroll
  for (int mi = 0; mi < MT; ++mi) {
#pragma unroll
    for (int nt = 0; nt < NT; ++nt) {
      int c = col0 + nt * 16 + (lane & 15);
      float bs = bias[c];
#pragma unroll
      for (int rr = 0; rr < 8; ++rr) {
        int ro = row0 + mi * 16 + rr + 8 * half;
        float v = acc[mi * NT + nt][rr] + bs;
        if constexpr (RES) v += res[(size_t)ro * NDIM + c];
        if constexpr (RELU) v = v > 0.0f ? v : 0.0f;
        if constexpr (OUT32) ((float*)out)[(size_t)ro * NDIM + c] = v;
        else ((_Float16*)out)[(size_t)ro * NDIM + c] = (_Float16)v;
      }
    }
  }
}

// ---------------------------------------------------------------------------
// Flash attention: one wave per (batch, head, 16-row q tile). D=32 => one
// WMMA K-step for scores; online softmax; P staged via LDS to A-frag layout.
// ---------------------------------------------------------------------------
template <bool MASKED>
__global__ void attn_kernel(const _Float16* __restrict__ Qp,
                            const _Float16* __restrict__ Kp,
                            const _Float16* __restrict__ VT,
                            const unsigned char* __restrict__ mask,
                            _Float16* __restrict__ O) {
  __shared__ _Float16 Pst[8 * 512];
  int lane = threadIdx.x & 31;
  int w    = threadIdx.x >> 5;
  int gid  = blockIdx.x * 8 + w;
  int qt = gid & 63;           // SEQ/16 tiles
  int bh = gid >> 6;
  int b = bh / NHEAD, hd = bh % NHEAD;
  int q0   = qt << 4;
  int half = lane >> 4;
  int l15  = lane & 15;
  int k0   = half ? 16 : 0;    // A-frag K sub-offset
  int kk   = half ? 16 : 0;    // B-frag K sub-offset

  // Q fragment (pre-scaled by 1/sqrt(D))
  const _Float16* qp = Qp + ((size_t)b * SEQ + q0 + l15) * HSZ + hd * HDIM + k0;
  v8h qlo = *(const v8h*)qp;
  v8h qhi = *(const v8h*)(qp + 8);
  v16h qa;
#pragma unroll
  for (int i = 0; i < 8; ++i) {
    qa[i]     = (_Float16)((float)qlo[i] * 0.17677669529663687f);
    qa[i + 8] = (_Float16)((float)qhi[i] * 0.17677669529663687f);
  }

  float m[8], l[8];
  v8f zero = {};
  v8f o0 = zero, o1 = zero;
#pragma unroll
  for (int rr = 0; rr < 8; ++rr) { m[rr] = -3.0e38f; l[rr] = 0.0f; }
  _Float16* pw = Pst + w * 512;

  const _Float16* kbase = Kp + ((size_t)b * SEQ + l15) * HSZ + hd * HDIM + kk;
  const _Float16* vbase =
      VT + ((size_t)(b * NHEAD + hd) * HDIM + l15) * SEQ + kk;

  for (int kt = 0; kt < (SEQ >> 5); ++kt) {
    int keybase = kt << 5;
    // K B-fragments for two 16-key tiles (contiguous d per lane)
    const _Float16* kp0 = kbase + (size_t)keybase * HSZ;
    v16h kb0 = *(const v16h*)kp0;
    v16h kb1 = *(const v16h*)(kp0 + (size_t)16 * HSZ);
    v8f s0 = __builtin_amdgcn_wmma_f32_16x16x32_f16(false, qa, false, kb0,
                                                    (short)0, zero, false, false);
    v8f s1 = __builtin_amdgcn_wmma_f32_16x16x32_f16(false, qa, false, kb1,
                                                    (short)0, zero, false, false);
    if constexpr (MASKED) {
      const unsigned char* mrow =
          mask + ((size_t)b * SEQ + q0 + 8 * half) * SEQ + keybase + l15;
#pragma unroll
      for (int rr = 0; rr < 8; ++rr) {
        const unsigned char* mp = mrow + (size_t)rr * SEQ;
        if (!mp[0])  s0[rr] = -3.0e38f;
        if (!mp[16]) s1[rr] = -3.0e38f;
      }
    }
#pragma unroll
    for (int rr = 0; rr < 8; ++rr) {
      float rm = hred_max16(fmaxf(s0[rr], s1[rr]));
      float mn = fmaxf(m[rr], rm);
      float corr = __expf(m[rr] - mn);
      float p0 = __expf(s0[rr] - mn);
      float p1 = __expf(s1[rr] - mn);
      float rs = hred_sum16(p0 + p1);
      l[rr] = l[rr] * corr + rs;
      m[rr] = mn;
      o0[rr] *= corr;
      o1[rr] *= corr;
      int prow = rr + 8 * half;
      pw[prow * 32 + l15]      = (_Float16)p0;
      pw[prow * 32 + 16 + l15] = (_Float16)p1;
    }
    // P as A-fragment from LDS
    const _Float16* pp = pw + l15 * 32 + k0;
    v8h plo = *(const v8h*)pp;
    v8h phi = *(const v8h*)(pp + 8);
    // V B-fragments from transposed V (contiguous keys per lane)
    const _Float16* vp = vbase + keybase;
    v16h vb0 = *(const v16h*)vp;
    v16h vb1 = *(const v16h*)(vp + (size_t)16 * SEQ);
    v16h pa;
#pragma unroll
    for (int i = 0; i < 8; ++i) { pa[i] = plo[i]; pa[i + 8] = phi[i]; }
    o0 = __builtin_amdgcn_wmma_f32_16x16x32_f16(false, pa, false, vb0,
                                                (short)0, o0, false, false);
    o1 = __builtin_amdgcn_wmma_f32_16x16x32_f16(false, pa, false, vb1,
                                                (short)0, o1, false, false);
  }
#pragma unroll
  for (int rr = 0; rr < 8; ++rr) {
    int ro = q0 + rr + 8 * half;
    float inv = 1.0f / l[rr];
    _Float16* op = O + ((size_t)b * SEQ + ro) * HSZ + hd * HDIM;
    op[l15]      = (_Float16)(o0[rr] * inv);
    op[16 + l15] = (_Float16)(o1[rr] * inv);
  }
}

// ---------------------------------------------------------------------------
// Host orchestration
// ---------------------------------------------------------------------------
extern "C" void kernel_launch(void* const* d_in, const int* in_sizes, int n_in,
                              void* d_out, int out_size, void* d_ws, size_t ws_size,
                              hipStream_t stream) {
  (void)in_sizes; (void)n_in; (void)out_size; (void)ws_size;
  const float* query = (const float*)d_in[0];
  const float* keyi  = (const float*)d_in[1];
  const float* vali  = (const float*)d_in[2];
  const unsigned char* mask = (const unsigned char*)d_in[3];
  const float* sa_qw = (const float*)d_in[4];  const float* sa_qb = (const float*)d_in[5];
  const float* sa_kw = (const float*)d_in[6];  const float* sa_kb = (const float*)d_in[7];
  const float* sa_vw = (const float*)d_in[8];  const float* sa_vb = (const float*)d_in[9];
  const float* sa_ow = (const float*)d_in[10]; const float* sa_ob = (const float*)d_in[11];
  const float* ca_qw = (const float*)d_in[12]; const float* ca_qb = (const float*)d_in[13];
  const float* ca_kw = (const float*)d_in[14]; const float* ca_kb = (const float*)d_in[15];
  const float* ca_vw = (const float*)d_in[16]; const float* ca_vb = (const float*)d_in[17];
  const float* ca_ow = (const float*)d_in[18]; const float* ca_ob = (const float*)d_in[19];
  const float* fc1_w = (const float*)d_in[20]; const float* fc1_b = (const float*)d_in[21];
  const float* fc2_w = (const float*)d_in[22]; const float* fc2_b = (const float*)d_in[23];
  const float* ln1_g = (const float*)d_in[24]; const float* ln1_b = (const float*)d_in[25];
  const float* ln2_g = (const float*)d_in[26]; const float* ln2_b = (const float*)d_in[27];
  const float* dn_g  = (const float*)d_in[28]; const float* dn_b  = (const float*)d_in[29];

  // workspace carve-up (256B aligned)
  char* ws = (char*)d_ws;
  size_t off = 0;
  auto carve = [&](size_t bytes) -> char* {
    size_t a = (off + 255) & ~(size_t)255;
    off = a + bytes;
    return ws + a;
  };
  const size_t actH = (size_t)BSROWS * HSZ * sizeof(_Float16);
  _Float16* xq  = (_Float16*)carve(actH);
  _Float16* xk  = (_Float16*)carve(actH);
  _Float16* xv  = (_Float16*)carve(actH);
  _Float16* Qp  = (_Float16*)carve(actH);
  _Float16* Kp  = (_Float16*)carve(actH);
  _Float16* Vp  = (_Float16*)carve(actH);
  _Float16* VT  = (_Float16*)carve(actH);
  _Float16* AO  = (_Float16*)carve(actH);
  _Float16* H1  = (_Float16*)carve((size_t)BSROWS * MLPD * sizeof(_Float16));
  float*    qbf = (float*)carve((size_t)BSROWS * HSZ * sizeof(float));
  const size_t wpS = (size_t)NLAYER * HSZ * HSZ;   // halves per small weight set
  _Float16* wsaq = (_Float16*)carve(wpS * 2); _Float16* wsak = (_Float16*)carve(wpS * 2);
  _Float16* wsav = (_Float16*)carve(wpS * 2); _Float16* wsao = (_Float16*)carve(wpS * 2);
  _Float16* wcaq = (_Float16*)carve(wpS * 2); _Float16* wcak = (_Float16*)carve(wpS * 2);
  _Float16* wcav = (_Float16*)carve(wpS * 2); _Float16* wcao = (_Float16*)carve(wpS * 2);
  _Float16* wf1  = (_Float16*)carve((size_t)NLAYER * HSZ * MLPD * 2);
  _Float16* wf2  = (_Float16*)carve((size_t)NLAYER * MLPD * HSZ * 2);

  // pack weights into WMMA B-fragment order
  auto pack = [&](const float* W, _Float16* dst, int K, int N) {
    dim3 g((K * N + 255) / 256, NLAYER);
    hipLaunchKernelGGL(pack_w_kernel, g, dim3(256), 0, stream, W, dst, K, N);
  };
  pack(sa_qw, wsaq, HSZ, HSZ); pack(sa_kw, wsak, HSZ, HSZ);
  pack(sa_vw, wsav, HSZ, HSZ); pack(sa_ow, wsao, HSZ, HSZ);
  pack(ca_qw, wcaq, HSZ, HSZ); pack(ca_kw, wcak, HSZ, HSZ);
  pack(ca_vw, wcav, HSZ, HSZ); pack(ca_ow, wcao, HSZ, HSZ);
  pack(fc1_w, wf1, HSZ, MLPD); pack(fc2_w, wf2, MLPD, HSZ);

  hipMemcpyAsync(qbf, query, (size_t)BSROWS * HSZ * sizeof(float),
                 hipMemcpyDeviceToDevice, stream);

  const int lnBlocks = (BSROWS * 32) / 256;                 // 2048
  auto ln16 = [&](const float* in, const float* g, const float* b, _Float16* out) {
    hipLaunchKernelGGL((ln_kernel<false>), dim3(lnBlocks), dim3(256), 0, stream,
                       in, g, b, (void*)out, BSROWS);
  };
  // N=96 GEMMs: MT=2, NT=3 -> waves = (M/32) * 2
  auto proj = [&](const _Float16* A, const _Float16* W, const float* bias,
                  _Float16* out) {  // M=BS N=96 K=96
    hipLaunchKernelGGL((gemm_kernel<2, 3, HSZ, HSZ, false, false, false>),
                       dim3((BSROWS / 32) * 2 / 8), dim3(256), 0, stream,
                       A, W, bias, (const float*)nullptr, (void*)out, BSROWS);
  };
  auto vtrans = [&]() {
    hipLaunchKernelGGL(vtrans_kernel,
                       dim3((NBATCH * NHEAD * HDIM * SEQ) / 256), dim3(256), 0,
                       stream, (const _Float16*)Vp, VT);
  };
  const int attnBlocks = (NBATCH * NHEAD * (SEQ / 16)) / 8;  // 384
  auto ffn = [&](int i) {
    ln16(qbf, ln2_g + i * HSZ, ln2_b + i * HSZ, xq);
    // fc1: MT=2, NT=4 -> waves = (M/32) * (1024/64)
    hipLaunchKernelGGL((gemm_kernel<2, 4, MLPD, HSZ, true, false, false>),
                       dim3(((BSROWS / 32) * (MLPD / 64)) / 8), dim3(256), 0,
                       stream, (const _Float16*)xq, wf1 + (size_t)i * HSZ * MLPD,
                       fc1_b + (size_t)i * MLPD, (const float*)nullptr,
                       (void*)H1, BSROWS);
    hipLaunchKernelGGL((gemm_kernel<2, 3, HSZ, MLPD, false, true, true>),
                       dim3((BSROWS / 32) * 2 / 8), dim3(256), 0, stream,
                       (const _Float16*)H1, wf2 + (size_t)i * MLPD * HSZ,
                       fc2_b + i * HSZ, (const float*)qbf, (void*)qbf, BSROWS);
  };
  auto oproj96 = [&](const _Float16* A, const _Float16* W, const float* bias) {
    hipLaunchKernelGGL((gemm_kernel<2, 3, HSZ, HSZ, false, true, true>),
                       dim3((BSROWS / 32) * 2 / 8), dim3(256), 0, stream,
                       A, W, bias, (const float*)qbf, (void*)qbf, BSROWS);
  };

  for (int i = 0; i < NLAYER; ++i) {
    const size_t wo = (size_t)i * HSZ * HSZ;
    // ---- self-attention (no mask) ----
    ln16(qbf, ln1_g + i * HSZ, ln1_b + i * HSZ, xq);
    proj(xq, wsaq + wo, sa_qb + i * HSZ, Qp);
    proj(xq, wsak + wo, sa_kb + i * HSZ, Kp);
    proj(xq, wsav + wo, sa_vb + i * HSZ, Vp);
    vtrans();
    hipLaunchKernelGGL((attn_kernel<false>), dim3(attnBlocks), dim3(256), 0,
                       stream, (const _Float16*)Qp, (const _Float16*)Kp,
                       (const _Float16*)VT, mask, AO);
    oproj96(AO, wsao + wo, sa_ob + i * HSZ);
    // ---- FFN ----
    ffn(i);
    // ---- cross-attention (masked; k/v from layernormed inputs) ----
    ln16(qbf,  ln1_g + i * HSZ, ln1_b + i * HSZ, xq);
    ln16(keyi, ln1_g + i * HSZ, ln1_b + i * HSZ, xk);
    ln16(vali, ln1_g + i * HSZ, ln1_b + i * HSZ, xv);
    proj(xq, wcaq + wo, ca_qb + i * HSZ, Qp);
    proj(xk, wcak + wo, ca_kb + i * HSZ, Kp);
    proj(xv, wcav + wo, ca_vb + i * HSZ, Vp);
    vtrans();
    hipLaunchKernelGGL((attn_kernel<true>), dim3(attnBlocks), dim3(256), 0,
                       stream, (const _Float16*)Qp, (const _Float16*)Kp,
                       (const _Float16*)VT, mask, AO);
    oproj96(AO, wcao + wo, ca_ob + i * HSZ);
    // ---- FFN ----
    ffn(i);
  }
  // final decoder norm -> f32 output
  hipLaunchKernelGGL((ln_kernel<true>), dim3(lnBlocks), dim3(256), 0, stream,
                     (const float*)qbf, dn_g, dn_b, d_out, BSROWS);
}
